// IA3MultiHeadAttention_20289425506491
// MI455X (gfx1250) — matrix operand order repaired and checked by reading, not compiled, not run on hardware
//
#include <hip/hip_runtime.h>

// ---------------------------------------------------------------------------
// Types for CDNA5 WMMA (wave32, 16x16x32 f16 -> f32)
// ---------------------------------------------------------------------------
typedef unsigned short u16;
typedef unsigned int   u32;
typedef __attribute__((ext_vector_type(4)))  u32   u32x4;
typedef __attribute__((ext_vector_type(8)))  float v8f;
typedef _Float16 f16;
typedef __attribute__((ext_vector_type(16))) f16   v16h;
typedef __attribute__((vector_size(16)))     int   i32x4v;  // builtin's pointee type

union FragH {            // 16 f16 elements = 8 VGPRs = one WMMA A/B fragment
    u32x4 q[2];
    v16h  h;
};

#define D_MODEL 1024
#define N_HEADS 16
#define D_HEAD  64
#define BB      2
#define LSEQ    2048
#define MROWS   (BB * LSEQ)   /* 4096 */

#define AS1 __attribute__((address_space(1)))
#define AS3 __attribute__((address_space(3)))

#if defined(__gfx1250__) && __has_builtin(__builtin_amdgcn_global_load_async_to_lds_b128)
#define HAVE_ASYNC_LDS 1
#endif

__device__ __forceinline__ u16 f2h_bits(float f) {
    union { f16 h; u16 u; } cv;
    cv.h = (f16)f;
    return cv.u;
}

__device__ __forceinline__ v8f wmma16(const FragH& a, const FragH& b, v8f c) {
    // emits v_wmma_f32_16x16x32_f16
    return __builtin_amdgcn_wmma_f32_16x16x32_f16(false, a.h, false, b.h,
                                                  (short)0, c, false, false);
}

// 16-byte global -> LDS copy: async (ASYNCcnt-tracked) on CDNA5, sync fallback
__device__ __forceinline__ void stage16(const u16* g, u16* l) {
#if defined(HAVE_ASYNC_LDS)
    __builtin_amdgcn_global_load_async_to_lds_b128(
        (AS1 i32x4v*)g, (AS3 i32x4v*)l, 0, 0);
#else
    *(u32x4*)l = *(const u32x4*)g;
#endif
}

__device__ __forceinline__ void async_wait() {
#if defined(HAVE_ASYNC_LDS)
#  if __has_builtin(__builtin_amdgcn_s_wait_asynccnt)
    __builtin_amdgcn_s_wait_asynccnt(0);
#  else
    asm volatile("s_wait_asynccnt 0x0" ::: "memory");
#  endif
#endif
}

// ---------------------------------------------------------------------------
// fp32 -> fp16 conversion, with optional per-row scale (IA3 fold into weights)
// ---------------------------------------------------------------------------
__global__ void cvt_f32_f16_kernel(const float* __restrict__ src,
                                   u16* __restrict__ dst,
                                   const float* __restrict__ scale,
                                   int rowlen, int n) {
    int i = blockIdx.x * blockDim.x + threadIdx.x;
    if (i >= n) return;
    float v = src[i];
    if (scale) v *= scale[i / rowlen];
    dst[i] = f2h_bits(v);
}

__global__ void scale_bias_kernel(const float* __restrict__ b,
                                  const float* __restrict__ s,
                                  float* __restrict__ dst, int n) {
    int i = blockIdx.x * blockDim.x + threadIdx.x;
    if (i < n) dst[i] = b[i] * s[i];
}

// ---------------------------------------------------------------------------
// GEMM: C[M,N] = A[M,K] @ W[N,K]^T + bias   (f16 in, f32 accumulate)
// Block = 8 waves sharing one 64-column weight panel; the 64x32 f16 weight
// K-slab is staged into LDS once per block (async, double-buffered).
// Each wave computes a 32x64 C tile: 2 A-frags x 4 B-frags = 8 WMMA / K-step.
// mode 0: f16 row-major out, mode 1: f16 per-head-transposed out (for V),
// mode 2: f32 row-major out (final projection -> d_out)
// ---------------------------------------------------------------------------
__global__ __launch_bounds__(256) void gemm_f16_wmma_kernel(
    const u16* __restrict__ A, const u16* __restrict__ W,
    const float* __restrict__ bias,
    u16* __restrict__ outH, float* __restrict__ outF,
    int M, int N, int K, int mode)
{
    __shared__ __align__(16) u16 Wt[2][64][32];   // 2 x 4KB double buffer

    const int tid  = threadIdx.x;
    const int lane = tid & 31;
    const int wave = tid >> 5;
    const int gw   = blockIdx.x * 8 + wave;
    const int mt32 = gw % (M >> 5);               // 32-row tile index
    const int nt   = gw / (M >> 5);               // shared by all waves in block
    const int col0 = nt * 64;
    const int m    = lane & 15;
    const int half = lane >> 4;

    // staging assignment: thread t copies W[col0 + (t>>2)][k0 + (t&3)*8 ..+7]
    const int srow = tid >> 2;
    const int skc  = (tid & 3) * 8;
    const u16* wsrc = W + (size_t)(col0 + srow) * K + skc;
    u16* wdst0 = &Wt[0][srow][skc];
    u16* wdst1 = &Wt[1][srow][skc];

    const u16* arow0 = A + (size_t)(mt32 * 32 + m) * K;        // rows 0..15
    const u16* arow1 = arow0 + (size_t)16 * K;                 // rows 16..31

    v8f acc[2][4] = {};

    stage16(wsrc, wdst0);
    async_wait();
    __syncthreads();

    int buf = 0;
    for (int k0 = 0; k0 < K; k0 += 32) {
        if (k0 + 32 < K) stage16(wsrc + k0 + 32, buf ? wdst0 : wdst1);

        FragH a0, a1;
        a0.q[0] = *(const u32x4*)(arow0 + k0 + 8 * half);
        a0.q[1] = *(const u32x4*)(arow0 + k0 + 16 + 8 * half);
        a1.q[0] = *(const u32x4*)(arow1 + k0 + 8 * half);
        a1.q[1] = *(const u32x4*)(arow1 + k0 + 16 + 8 * half);

        #pragma unroll
        for (int f = 0; f < 4; ++f) {
            FragH b;
            const u16* wl = &Wt[buf][f * 16 + m][8 * half];
            b.q[0] = *(const u32x4*)(wl);
            b.q[1] = *(const u32x4*)(wl + 16);
            acc[0][f] = wmma16(a0, b, acc[0][f]);
            acc[1][f] = wmma16(a1, b, acc[1][f]);
        }

        async_wait();
        __syncthreads();
        buf ^= 1;
    }

    #pragma unroll
    for (int t = 0; t < 2; ++t) {
        #pragma unroll
        for (int f = 0; f < 4; ++f) {
            const int col = col0 + f * 16 + m;
            const float bc = bias ? bias[col] : 0.f;
            #pragma unroll
            for (int v = 0; v < 8; ++v) {
                const int row = mt32 * 32 + t * 16 + v + 8 * half;
                const float val = acc[t][f][v] + bc;
                if (mode == 0) {
                    outH[(size_t)row * N + col] = f2h_bits(val);
                } else if (mode == 1) {
                    // V output transposed per head: Vt[b][h][d][l]
                    const int bb = row >> 11, l = row & (LSEQ - 1);
                    const int hh = col >> 6,  d = col & (D_HEAD - 1);
                    outH[(((size_t)(bb * N_HEADS + hh)) * D_HEAD + d) * LSEQ + l] = f2h_bits(val);
                } else {
                    outF[(size_t)row * N + col] = val;
                }
            }
        }
    }
}

// ---------------------------------------------------------------------------
// Flash-style causal attention. Block = 8 waves sharing (b,h), adjacent
// 16-query tiles. K (32x64) and V (64x32, pre-transposed) blocks are staged
// into LDS once per block (async double-buffered) and consumed by all waves.
// Online softmax over 32-key blocks; scores and P.V via WMMA.
// ---------------------------------------------------------------------------
__global__ __launch_bounds__(256) void attn_f16_wmma_kernel(
    const u16* __restrict__ Q, const u16* __restrict__ Km,
    const u16* __restrict__ Vt, u16* __restrict__ ctx)
{
    __shared__ __align__(16) u16 Kt[2][32][64];      // 2 x 4KB: [key][d]
    __shared__ __align__(16) u16 Vl[2][64][32];      // 2 x 4KB: [d][key]
    __shared__ __align__(16) u16 plds[8][16 * 32];   // per-wave P staging

    const int tid  = threadIdx.x;
    const int lane = tid & 31;
    const int wave = tid >> 5;
    const int bh   = blockIdx.x >> 4;       // 0..31  (same for whole block)
    const int qb   = blockIdx.x & 15;       // q-tile octet
    const int b    = bh >> 4;
    const int h    = bh & 15;
    const int qt   = qb * 8 + wave;
    const int q0   = qt * 16;
    const int m    = lane & 15;
    const int half = lane >> 4;
    const float CSCALE = 0.125f * 1.44269504088896340736f; // 1/sqrt(64)*log2(e)

    const u16* kbase = Km + (size_t)(b * LSEQ) * D_MODEL + h * D_HEAD;
    const u16* vbase = Vt + ((size_t)(b * N_HEADS + h)) * D_HEAD * LSEQ;

    // staging assignments (per thread, 16B each for K and V)
    const int krow = tid >> 3;                 // 0..31 key
    const int kdc  = (tid & 7) * 8;            // d chunk
    const int vrow = tid >> 2;                 // 0..63 d
    const int vlc  = (tid & 3) * 8;            // key chunk
    const u16* ksrc = kbase + (size_t)krow * D_MODEL + kdc;
    const u16* vsrc = vbase + (size_t)vrow * LSEQ + vlc;

    // Q A-fragments for d = 0..31 and d = 32..63
    const u16* qrow = Q + (size_t)(b * LSEQ + q0 + m) * D_MODEL + h * D_HEAD;
    FragH qa0, qa1;
    qa0.q[0] = *(const u32x4*)(qrow + 8 * half);
    qa0.q[1] = *(const u32x4*)(qrow + 16 + 8 * half);
    qa1.q[0] = *(const u32x4*)(qrow + 32 + 8 * half);
    qa1.q[1] = *(const u32x4*)(qrow + 48 + 8 * half);

    v8f o0 = {}, o1 = {}, o2 = {}, o3 = {};
    float mrow[8], lrow[8];
    #pragma unroll
    for (int v = 0; v < 8; ++v) { mrow[v] = -1e30f; lrow[v] = 0.f; }

    u16* pl = &plds[wave][0];

    const int myblocks = ((q0 + 15) >> 5) + 1;             // this wave's causal range
    const int nbmax    = ((qb * 128 + 112 + 15) >> 5) + 1; // largest in block

    stage16(ksrc, &Kt[0][krow][kdc]);
    stage16(vsrc, &Vl[0][vrow][vlc]);
    async_wait();
    __syncthreads();

    int buf = 0;
    for (int blk = 0; blk < nbmax; ++blk) {
        const int j0 = blk * 32;
        if (blk + 1 < nbmax) {
            const int nb = buf ^ 1;
            stage16(ksrc + (size_t)32 * D_MODEL, &Kt[nb][krow][kdc]);
            stage16(vsrc + 32,                   &Vl[nb][vrow][vlc]);
            ksrc += (size_t)32 * D_MODEL;
            vsrc += 32;
        }

        if (blk < myblocks) {   // wave-uniform: EXEC stays full for WMMA
            // ---- scores: two 16x16 tiles (keys j0..+15, j0+16..+31) ----
            v8f s0v = {}, s1v = {};
            {
                FragH kb;
                const u16* k0p = &Kt[buf][m][8 * half];
                kb.q[0] = *(const u32x4*)(k0p);      kb.q[1] = *(const u32x4*)(k0p + 16);
                s0v = wmma16(qa0, kb, s0v);
                kb.q[0] = *(const u32x4*)(k0p + 32); kb.q[1] = *(const u32x4*)(k0p + 48);
                s0v = wmma16(qa1, kb, s0v);

                const u16* k1p = &Kt[buf][16 + m][8 * half];
                kb.q[0] = *(const u32x4*)(k1p);      kb.q[1] = *(const u32x4*)(k1p + 16);
                s1v = wmma16(qa0, kb, s1v);
                kb.q[0] = *(const u32x4*)(k1p + 32); kb.q[1] = *(const u32x4*)(k1p + 48);
                s1v = wmma16(qa1, kb, s1v);
            }

            // ---- online softmax update over this 32-key block ----
            const bool fullblk = (j0 + 31 <= q0);
            float p0[8], p1[8], alpha[8];
            #pragma unroll
            for (int v = 0; v < 8; ++v) {
                const int qpos = q0 + v + 8 * half;
                float s0 = s0v[v] * CSCALE;
                float s1 = s1v[v] * CSCALE;
                if (!fullblk) {
                    if (j0 + m > qpos)      s0 = -1e30f;   // causal mask
                    if (j0 + 16 + m > qpos) s1 = -1e30f;
                }
                float mv = fmaxf(s0, s1);                  // row max (16 lanes/half)
                mv = fmaxf(mv, __shfl_xor(mv, 1, 32));
                mv = fmaxf(mv, __shfl_xor(mv, 2, 32));
                mv = fmaxf(mv, __shfl_xor(mv, 4, 32));
                mv = fmaxf(mv, __shfl_xor(mv, 8, 32));
                const float mnew = fmaxf(mrow[v], mv);
                alpha[v] = __builtin_exp2f(mrow[v] - mnew);
                p0[v]    = __builtin_exp2f(s0 - mnew);
                p1[v]    = __builtin_exp2f(s1 - mnew);
                float rs = p0[v] + p1[v];                  // row sum
                rs += __shfl_xor(rs, 1, 32);
                rs += __shfl_xor(rs, 2, 32);
                rs += __shfl_xor(rs, 4, 32);
                rs += __shfl_xor(rs, 8, 32);
                lrow[v] = lrow[v] * alpha[v] + rs;
                mrow[v] = mnew;
            }

            // rescale accumulators, stage P (C-layout -> LDS row-major f16)
            #pragma unroll
            for (int v = 0; v < 8; ++v) {
                o0[v] *= alpha[v]; o1[v] *= alpha[v];
                o2[v] *= alpha[v]; o3[v] *= alpha[v];
                const int r = v + 8 * half;
                pl[r * 32 + m]      = f2h_bits(p0[v]);
                pl[r * 32 + 16 + m] = f2h_bits(p1[v]);
            }
            __asm__ volatile("" ::: "memory");   // keep DS store->load ordered

            FragH pa;
            pa.q[0] = *(const u32x4*)(pl + m * 32 + 8 * half);
            pa.q[1] = *(const u32x4*)(pl + m * 32 + 16 + 8 * half);

            // ---- O += P @ V  (B frags from LDS-staged transposed V) ----
            FragH vb;
            const u16* v0p = &Vl[buf][m][8 * half];
            vb.q[0] = *(const u32x4*)(v0p);          vb.q[1] = *(const u32x4*)(v0p + 16);
            o0 = wmma16(pa, vb, o0);
            const u16* v1p = &Vl[buf][16 + m][8 * half];
            vb.q[0] = *(const u32x4*)(v1p);          vb.q[1] = *(const u32x4*)(v1p + 16);
            o1 = wmma16(pa, vb, o1);
            const u16* v2p = &Vl[buf][32 + m][8 * half];
            vb.q[0] = *(const u32x4*)(v2p);          vb.q[1] = *(const u32x4*)(v2p + 16);
            o2 = wmma16(pa, vb, o2);
            const u16* v3p = &Vl[buf][48 + m][8 * half];
            vb.q[0] = *(const u32x4*)(v3p);          vb.q[1] = *(const u32x4*)(v3p + 16);
            o3 = wmma16(pa, vb, o3);
        }

        async_wait();
        __syncthreads();
        buf ^= 1;
    }

    // ---- epilogue: normalize and write ctx (f16) ----
    float inv[8];
    #pragma unroll
    for (int v = 0; v < 8; ++v) inv[v] = 1.f / lrow[v];
    #pragma unroll
    for (int f = 0; f < 4; ++f) {
        v8f ov = (f == 0) ? o0 : (f == 1) ? o1 : (f == 2) ? o2 : o3;
        #pragma unroll
        for (int v = 0; v < 8; ++v) {
            const int row = b * LSEQ + q0 + v + 8 * half;
            const int col = h * D_HEAD + f * 16 + m;
            ctx[(size_t)row * D_MODEL + col] = f2h_bits(ov[v] * inv[v]);
        }
    }
}

// ---------------------------------------------------------------------------
// Host-side orchestration
// ---------------------------------------------------------------------------
extern "C" void kernel_launch(void* const* d_in, const int* in_sizes, int n_in,
                              void* d_out, int out_size, void* d_ws, size_t ws_size,
                              hipStream_t stream) {
    const float* x    = (const float*)d_in[0];
    // d_in[1] = causal_mask (unused: causality applied analytically)
    const float* Wq   = (const float*)d_in[2];
    const float* bq   = (const float*)d_in[3];
    const float* Wk   = (const float*)d_in[4];
    const float* bk   = (const float*)d_in[5];
    const float* Wv   = (const float*)d_in[6];
    const float* bv   = (const float*)d_in[7];
    const float* Wo   = (const float*)d_in[8];
    const float* bo   = (const float*)d_in[9];
    const float* ia3k = (const float*)d_in[10];
    const float* ia3v = (const float*)d_in[11];

    char* ws = (char*)d_ws;
    const size_t MB = 1024u * 1024u;
    u16*   XB  = (u16*)(ws + 0 * MB);    // 8 MB   x  in f16
    u16*   WQB = (u16*)(ws + 8 * MB);    // 2 MB
    u16*   WKB = (u16*)(ws + 10 * MB);   // 2 MB (ia3_k folded)
    u16*   WVB = (u16*)(ws + 12 * MB);   // 2 MB (ia3_v folded)
    u16*   WOB = (u16*)(ws + 14 * MB);   // 2 MB
    u16*   QB  = (u16*)(ws + 16 * MB);   // 8 MB
    u16*   KB  = (u16*)(ws + 24 * MB);   // 8 MB
    u16*   VT  = (u16*)(ws + 32 * MB);   // 8 MB  V transposed per head
    u16*   CTX = (u16*)(ws + 40 * MB);   // 8 MB  attention context
    float* BK2 = (float*)(ws + 48 * MB); // 4 KB  bk * ia3_k
    float* BV2 = (float*)(ws + 48 * MB + 8192);

    const int nX = MROWS * D_MODEL;       // 4,194,304
    const int nW = D_MODEL * D_MODEL;     // 1,048,576

    cvt_f32_f16_kernel<<<(nX + 255) / 256, 256, 0, stream>>>(x,  XB,  nullptr, D_MODEL, nX);
    cvt_f32_f16_kernel<<<(nW + 255) / 256, 256, 0, stream>>>(Wq, WQB, nullptr, D_MODEL, nW);
    cvt_f32_f16_kernel<<<(nW + 255) / 256, 256, 0, stream>>>(Wk, WKB, ia3k,    D_MODEL, nW);
    cvt_f32_f16_kernel<<<(nW + 255) / 256, 256, 0, stream>>>(Wv, WVB, ia3v,    D_MODEL, nW);
    cvt_f32_f16_kernel<<<(nW + 255) / 256, 256, 0, stream>>>(Wo, WOB, nullptr, D_MODEL, nW);
    scale_bias_kernel<<<4, 256, 0, stream>>>(bk, ia3k, BK2, D_MODEL);
    scale_bias_kernel<<<4, 256, 0, stream>>>(bv, ia3v, BV2, D_MODEL);

    // (MROWS/32) * (D_MODEL/64) waves / 8 waves-per-block = 256 blocks
    const int gemmBlocks = (MROWS / 32) * (D_MODEL / 64) / 8;
    gemm_f16_wmma_kernel<<<gemmBlocks, 256, 0, stream>>>(XB, WQB, bq,  QB, nullptr, MROWS, D_MODEL, D_MODEL, 0);
    gemm_f16_wmma_kernel<<<gemmBlocks, 256, 0, stream>>>(XB, WKB, BK2, KB, nullptr, MROWS, D_MODEL, D_MODEL, 0);
    gemm_f16_wmma_kernel<<<gemmBlocks, 256, 0, stream>>>(XB, WVB, BV2, VT, nullptr, MROWS, D_MODEL, D_MODEL, 1);

    // 512 blocks: 32 (b,h) pairs x 16 q-tile octets
    attn_f16_wmma_kernel<<<BB * N_HEADS * (LSEQ / 16 / 8), 256, 0, stream>>>(QB, KB, VT, CTX);

    gemm_f16_wmma_kernel<<<gemmBlocks, 256, 0, stream>>>(CTX, WOB, bo, nullptr, (float*)d_out,
                                                         MROWS, D_MODEL, D_MODEL, 2);
}
